// NeuralCA_23364622090406
// MI455X (gfx1250) — matrix operand order repaired and checked by reading, not compiled, not run on hardware
//
#include <hip/hip_runtime.h>
#include <stdint.h>

typedef __attribute__((ext_vector_type(16))) __bf16 v16bf;
typedef __attribute__((ext_vector_type(8)))  __bf16 v8bf;
typedef __attribute__((ext_vector_type(2)))  __bf16 bf16x2;
typedef __attribute__((ext_vector_type(8)))  float  v8f;

#define HH    256
#define WW    256
#define C_IN  16
#define C_MID 128
#define K3    144        // 16 * 3 * 3
#define KC3   5          // ceil(144/32) K-chunks for conv3x3 (padded to 160)
#define KC1   4          // 128/32 K-chunks for conv1x1
#define TW    264        // padded LDS row width: x = -1..256 -> idx 0..257, pad to 264
#define HROWS (3 * C_IN + 1)       // 48 halo rows + 1 zero row for K padding
#define ZROW  (3 * C_IN * TW)      // flat offset of the zero row
#define PPAD  40                   // p_s chunk row stride in halves (32 + 8 -> 80 B)

// A-matrix (16x32 bf16) K index for vgpr v, lane-half hl, lo half of dword
__host__ __device__ constexpr int kkA(int v, int hl, int lo) {
    return hl * 16 + v * 2 + lo;
}
// B-matrix (32x16 bf16): lanes 0-15 hold K=0..15, lanes 16-31 hold K=16..31
// im2col LDS offset for absolute K (k = ci*9 + ky*3 + kx); padded K -> zero row
__host__ __device__ constexpr int offB(int k) {
    return (k < K3) ? (((k % 9) / 3) * C_IN + (k / 9)) * TW + (k % 9) % 3
                    : ZROW;
}

// ---------------- one-time weight repack: fp32 -> bf16 fragment order ----------
__global__ __launch_bounds__(256)
void prep_weights(const float* __restrict__ w1, const float* __restrict__ w2,
                  __bf16* __restrict__ w1f, __bf16* __restrict__ w2f)
{
    int i = blockIdx.x * 256 + threadIdx.x;
    if (i < 8 * KC3 * 32 * 16) {               // w1f[mt][kc][lane][e]
        int e = i & 15, lane = (i >> 4) & 31, rest = i >> 9;
        int kc = rest % KC3, mt = rest / KC3;
        int hl = lane >> 4, ln = lane & 15;
        int kk = kc * 32 + kkA(e >> 1, hl, e & 1);
        float wv = (kk < K3) ? w1[(mt * 16 + ln) * K3 + kk] : 0.f;
        w1f[i] = (__bf16)wv;
    }
    if (i < KC1 * 32 * 16) {                   // w2f[kc][lane][e]
        int e = i & 15, lane = (i >> 4) & 31, kc = i >> 9;
        int hl = lane >> 4, ln = lane & 15;
        int kk = kc * 32 + kkA(e >> 1, hl, e & 1);
        w2f[i] = (__bf16)w2[ln * C_MID + kk];
    }
}

// ---------------- fused NCA step: conv3x3 -> relu -> conv1x1 -> add -> clamp ---
__global__ __launch_bounds__(256)
void nca_step_kernel(const float* __restrict__ hin,
                     float*       __restrict__ hout,
                     const v16bf* __restrict__ w1v,   // [mt*KC3+kc][lane]
                     const v16bf* __restrict__ w2v)   // [kc][lane]
{
    __shared__ __align__(16) __bf16 h_s[HROWS * TW];       // 25872 B
    __shared__ __align__(16) __bf16 p_s[8][2][16][PPAD];   // 20480 B (K-chunk buf)

    const int tid  = threadIdx.x;
    const int lane = tid & 31;
    const int wave = tid >> 5;
    const int hl   = lane >> 4;
    const int ln   = lane & 15;

    const int b = blockIdx.x >> 8;     // batch
    const int y = blockIdx.x & 255;    // output row

    // ---- stage halo rows (bf16 pairs) + zero row for K padding ----
    for (int idx = tid; idx < HROWS * (TW / 2); idx += 256) {
        int xp = (idx % (TW / 2)) * 2;
        int t  = idx / (TW / 2);
        float v0 = 0.f, v1 = 0.f;
        if (t < 3 * C_IN) {
            int c = t % C_IN, r = t / C_IN;
            int yy = y + r - 1;
            if (yy >= 0 && yy < HH) {
                const float* rowp = hin + ((size_t)(b * C_IN + c) * HH + yy) * WW;
                if (xp >= 1 && xp <= WW) v0 = rowp[xp - 1];   // xg = xp-1
                if (xp < WW)             v1 = rowp[xp];       // xg = xp
            }
        }
        bf16x2 pr = { (__bf16)v0, (__bf16)v1 };
        *(bf16x2*)&h_s[t * TW + xp] = pr;                     // ds_store_b32
    }
    __syncthreads();

    // ---- B fragments for BOTH of this wave's 16-pixel N-tiles ----
    v16bf bfr[2][KC3];
#pragma unroll
    for (int tt = 0; tt < 2; ++tt) {
        const int xb = (wave * 2 + tt) * 16 + ln;
#pragma unroll
        for (int kc = 0; kc < KC3; ++kc) {
#pragma unroll
            for (int e = 0; e < 16; ++e) {
                const int k0 = kc * 32 + (e >> 1) * 2 + (e & 1);  // hl == 0
                const int o0 = offB(k0);
                const int o1 = offB(k0 + 16);                     // hl == 1
                bfr[tt][kc][e] = h_s[(hl ? o1 : o0) + xb];        // ds_load_u16
            }
        }
    }

    // ---- fused GEMM1 (K=160) + interleaved GEMM2 (K=128, one chunk per mtp) --
    v8f acc2[2] = { {}, {} };
#pragma unroll 1
    for (int mtp = 0; mtp < 4; ++mtp) {        // 2 mt groups = one 32-ch K-chunk
#pragma unroll
        for (int sub = 0; sub < 2; ++sub) {
            const int mt = mtp * 2 + sub;
            v16bf afrag[KC3];
#pragma unroll
            for (int kc = 0; kc < KC3; ++kc)
                afrag[kc] = w1v[(mt * KC3 + kc) * 32 + lane];     // 2x b128 each
#pragma unroll
            for (int tt = 0; tt < 2; ++tt) {                     // reuse A for 2 tiles
                v8f acc = {};
#pragma unroll
                for (int kc = 0; kc < KC3; ++kc)
                    acc = __builtin_amdgcn_wmma_f32_16x16x32_bf16(
                        false, afrag[kc], false, bfr[tt][kc], (short)0, acc, false, false);
                v8bf pk;
#pragma unroll
                for (int r = 0; r < 8; ++r)
                    pk[r] = (__bf16)fmaxf(acc[r], 0.f);          // ReLU, pk-cvt
                // local channel = sub*16 + r + 8*hl  (chunk holds 32 channels)
                *(v8bf*)&p_s[wave][tt][ln][sub * 16 + 8 * hl] = pk;  // ds_store_b128
            }
        }
        // one full 32-channel chunk of relu(p) ready for both tiles
        v16bf a2 = w2v[mtp * 32 + lane];
#pragma unroll
        for (int tt = 0; tt < 2; ++tt) {
            v16bf b2 = *(const v16bf*)&p_s[wave][tt][ln][hl * 16];   // 2x b128
            acc2[tt] = __builtin_amdgcn_wmma_f32_16x16x32_bf16(
                false, a2, false, b2, (short)0, acc2[tt], false, false);
        }
        // per-wave LDS region + same-wave DS ordering: no barrier needed
    }

    // ---- h_new = clamp(h + dx) in full fp32 ----
#pragma unroll
    for (int tt = 0; tt < 2; ++tt) {
        const int x0 = (wave * 2 + tt) * 16;
#pragma unroll
        for (int r = 0; r < 8; ++r) {
            int ch = r + 8 * hl;
            size_t gi = ((size_t)(b * C_IN + ch) * HH + y) * WW + x0 + ln;
            float hv = hin[gi] + acc2[tt][r];
            float lo = (ch == 0) ? 0.f : -3.f;
            float hi = (ch == 0) ? 1.f :  3.f;
            hout[gi] = fminf(fmaxf(hv, lo), hi);
        }
    }
}

extern "C" void kernel_launch(void* const* d_in, const int* in_sizes, int n_in,
                              void* d_out, int out_size, void* d_ws, size_t ws_size,
                              hipStream_t stream) {
    const float* x  = (const float*)d_in[0];   // (8,16,256,256)
    const float* w1 = (const float*)d_in[1];   // (128,16,3,3)
    const float* w2 = (const float*)d_in[2];   // (16,128,1,1)
    float* out = (float*)d_out;

    const size_t STATE = (size_t)8 * C_IN * HH * WW;      // 8,388,608 floats
    float*  ws  = (float*)d_ws;                            // ping buffer (33.5 MB)
    __bf16* w1f = (__bf16*)((char*)d_ws + STATE * sizeof(float));
    __bf16* w2f = w1f + 8 * KC3 * 32 * 16;

    // one-time weight repack into WMMA fragment order (bf16)
    hipLaunchKernelGGL(prep_weights, dim3(80), dim3(256), 0, stream, w1, w2, w1f, w2f);

    const int steps = 64;                                  // matches reference setup
    dim3 grid(8 * 256), block(256);
    // ping-pong: even steps write ws, odd steps write d_out; step 63 -> d_out
    for (int s = 0; s < steps; ++s) {
        const float* src = (s == 0) ? x : ((s & 1) ? ws : out);
        float* dst = (s & 1) ? out : ws;
        hipLaunchKernelGGL(nca_step_kernel, grid, block, 0, stream,
                           src, dst, (const v16bf*)w1f, (const v16bf*)w2f);
    }
}
